// dissected_Conv2d_32650341384236
// MI455X (gfx1250) — matrix operand order repaired and checked by reading, not compiled, not run on hardware
//
#include <hip/hip_runtime.h>

typedef __attribute__((ext_vector_type(16))) __bf16 v16bf;
typedef __attribute__((ext_vector_type(8)))  __bf16 v8bf;
typedef __attribute__((ext_vector_type(8)))  float  v8f;
typedef __attribute__((ext_vector_type(4)))  unsigned int u32x4;
typedef __attribute__((ext_vector_type(8)))  int          i32x8;
typedef __attribute__((ext_vector_type(4)))  int          i32x4;

#define IN_C  64
#define OUT_C 64
#define HH    56
#define WW    56

// Workspace layout (bf16):
//   wprep: [kh][kw][o][i]              = 9*64*64  elems = 73728 B
//   xprep: [b][hp(58)][wp(58)][i(64)]  = 8*58*58*64 elems, zero-padded borders
#define WPREP_ELEMS (9 * 64 * 64)
#define XPREP_OFF_B (WPREP_ELEMS * 2)          // 73728, 128B aligned

// LDS layout (bytes): TDM writes 128B of data + 16B pad -> i-stride 72 elems
//   wlds: 3*64 rows * 72 bf16 = 27648
//   xlds: 4*58 rows * 72 bf16 = 33408
//   blds: 64 f32              =   256
#define XLDS_OFF   27648
#define BLDS_OFF   61056
#define SMEM_BYTES 61312

// ---------------- prep kernels ----------------

__global__ __launch_bounds__(256)
void prep_weights(const float* __restrict__ wgt, __bf16* __restrict__ wp)
{
    int d = blockIdx.x * 256 + threadIdx.x;      // dest: ((kh*3+kw)*64+o)*64+i
    if (d < WPREP_ELEMS) {
        int i = d & 63;
        int o = (d >> 6) & 63;
        int t = d >> 12;                          // t = kh*3+kw
        wp[d] = (__bf16)wgt[(o * IN_C + i) * 9 + t];
    }
}

__global__ __launch_bounds__(256)
void prep_x(const float* __restrict__ x, __bf16* __restrict__ xp)
{
    __shared__ __bf16 tile[WW * 64];              // [w][i]
    const int b  = blockIdx.x / 58;
    const int hp = blockIdx.x % 58;
    const int h  = hp - 1;
    const bool valid = (h >= 0 && h < HH);
    if (valid) {
        for (int idx = threadIdx.x; idx < 64 * WW; idx += 256) {
            int w = idx % WW;
            int i = idx / WW;                     // coalesced in w
            tile[w * 64 + i] = (__bf16)x[((b * IN_C + i) * HH + h) * WW + w];
        }
    }
    __syncthreads();
    const __bf16 z = (__bf16)0.0f;
    for (int idx = threadIdx.x; idx < 58 * 64; idx += 256) {
        int i  = idx & 63;
        int wp = idx >> 6;
        __bf16 v = z;
        if (valid && wp >= 1 && wp <= WW) v = tile[(wp - 1) * 64 + i];
        xp[((size_t)(b * 58 + hp) * 58 + wp) * 64 + i] = v;   // coalesced
    }
}

// ---------------- TDM 1D staged copy (clang-23 6-arg builtin) ----------------
// D# group0/group1 packed per CDNA5 ISA:
// count=1, type=2; data_size=1 (2B); pad_enable=1, pad_interval=4 (32 DWORDs),
// pad_amount=3 (4 DWORDs): +16B pad after every 128B -> LDS stride 72 bf16.
__device__ __forceinline__
void tdm_load_1d(unsigned lds_off, const void* gptr, unsigned nelem)
{
    unsigned long long ga = (unsigned long long)(uintptr_t)gptr;
    u32x4 g0;
    g0[0] = 1u;                                              // count=1
    g0[1] = lds_off;                                         // lds_addr (bytes)
    g0[2] = (unsigned)(ga & 0xFFFFFFFFull);                  // global_addr lo
    g0[3] = (unsigned)((ga >> 32) & 0x1FFFFFFull) | (2u << 30); // hi | type=2
    i32x8 g1;
    g1[0] = (1 << 16) | (1 << 20) | (4 << 22) | (3 << 25);   // dsz|pad en/int/amt
    g1[1] = (int)((nelem & 0xFFFFu) << 16);                  // tensor_dim0 lo
    g1[2] = (int)((nelem >> 16) | (1u << 16));               // dim0 hi | dim1=1
    g1[3] = (int)(nelem << 16);                              // tile_dim0
    g1[4] = 0;                                               // tile_dim1/2 = 0
    g1[5] = (int)nelem;                                      // dim0_stride lo
    g1[6] = 0;
    g1[7] = 0;
    i32x4 gz4 = (i32x4)0;                                    // groups 2/3: NULL
    i32x8 gz8 = (i32x8)0;                                    // unused group
    __builtin_amdgcn_tensor_load_to_lds(g0, g1, gz4, gz4, gz8, 0);
}

// ---------------- main kernel ----------------

__global__ __launch_bounds__(256)
void conv3x3_wmma_tdm(const __bf16* __restrict__ wprep,
                      const __bf16* __restrict__ xprep,
                      const float* __restrict__ bias,
                      float* __restrict__ out)
{
    extern __shared__ char smem[];
    __bf16* wlds = (__bf16*)smem;
    __bf16* xlds = (__bf16*)(smem + XLDS_OFF);
    float*  blds = (float*)(smem + BLDS_OFF);

    const int tid   = threadIdx.x;
    const int bi    = blockIdx.x / 28;
    const int hbase = (blockIdx.x % 28) * 2;

    if (tid >= 32 && tid < 96) blds[tid - 32] = bias[tid - 32];

    if (tid < 32) {   // wave 0 drives the Tensor Data Mover
        tdm_load_1d((unsigned)(unsigned long long)(uintptr_t)xlds,
                    xprep + ((size_t)(bi * 58 + hbase) * 58) * 64, 4 * 58 * 64);
        tdm_load_1d((unsigned)(unsigned long long)(uintptr_t)wlds,
                    wprep, 3 * 64 * 64);
        __builtin_amdgcn_s_wait_tensorcnt(0);
    }
    __syncthreads();

    // 8 waves: nt = wave>>1 (N tile), mp = wave&1 (pair of M tiles)
    const int wave = tid >> 5;
    const int lane = tid & 31;
    const int ln   = lane & 15;
    const int half = lane >> 4;
    const int nt   = wave >> 1;
    const int mp   = wave & 1;
    int n0 = nt * 16;
    if (nt == 3) n0 = 40;                 // overlap last tile: all n < 56
    const int n  = n0 + ln;
    const int m0 = mp * 32 + ln;          // A row, M tiles m0 and m0+16

    v8f acc[2][2] = {};                   // [row rr][m tile mi]

    for (int g = 0; g < 3; ++g) {         // g == kh
        if (g) {
            __syncthreads();              // all readers done with wlds
            if (tid < 32) {
                tdm_load_1d((unsigned)(unsigned long long)(uintptr_t)wlds,
                            wprep + g * (3 * 64 * 64), 3 * 64 * 64);
                __builtin_amdgcn_s_wait_tensorcnt(0);
            }
            __syncthreads();
        }
        if (g < 2)                        // warm next group for the TDM (GL2)
            __builtin_prefetch(wprep + (g + 1) * (3 * 64 * 64) + tid * 48, 0, 0);

        #pragma unroll
        for (int kw = 0; kw < 3; ++kw) {
            #pragma unroll
            for (int kc = 0; kc < 2; ++kc) {
                // A 16x32 bf16: lane K = half*8 + j (j<8), 16+half*8+(j-8)
                const __bf16* a0p = &wlds[(kw * 64 + m0) * 72 + kc * 32 + half * 8];
                const __bf16* a1p = a0p + 16 * 72;
                v8bf a0l = *(const v8bf*)(a0p);
                v8bf a0h = *(const v8bf*)(a0p + 16);
                v8bf a1l = *(const v8bf*)(a1p);
                v8bf a1h = *(const v8bf*)(a1p + 16);
                v16bf va0, va1;
                #pragma unroll
                for (int j = 0; j < 8; ++j) {
                    va0[j] = a0l[j]; va0[j + 8] = a0h[j];
                    va1[j] = a1l[j]; va1[j + 8] = a1h[j];
                }
                #pragma unroll
                for (int rr = 0; rr < 2; ++rr) {
                    // B 32x16 bf16: lane K = half*8 + j (j<8), 16+half*8+(j-8)
                    const __bf16* bp =
                        &xlds[((rr + g) * 58 + (n + kw)) * 72 + kc * 32 + half * 8];
                    v8bf b0 = *(const v8bf*)(bp);
                    v8bf b1 = *(const v8bf*)(bp + 16);
                    v16bf vb;
                    #pragma unroll
                    for (int j = 0; j < 8; ++j) { vb[j] = b0[j]; vb[j + 8] = b1[j]; }
                    acc[rr][0] = __builtin_amdgcn_wmma_f32_16x16x32_bf16(
                        false, va0, false, vb, (short)0, acc[rr][0], false, false);
                    acc[rr][1] = __builtin_amdgcn_wmma_f32_16x16x32_bf16(
                        false, va1, false, vb, (short)0, acc[rr][1], false, false);
                }
            }
        }
    }

    // C/D: VGPR p -> M = p (lanes 0-15) / p+8 (lanes 16-31), N = ln.
    if (!(nt == 2 && ln >= 8)) {          // skip columns duplicated by tile 3
        #pragma unroll
        for (int rr = 0; rr < 2; ++rr) {
            const int h = hbase + rr;
            #pragma unroll
            for (int mi = 0; mi < 2; ++mi) {
                #pragma unroll
                for (int p = 0; p < 8; ++p) {
                    const int o = mp * 32 + mi * 16 + half * 8 + p;
                    out[((bi * OUT_C + o) * HH + h) * WW + n] = acc[rr][mi][p] + blds[o];
                }
            }
        }
    }
}

// ---------------- host ----------------

extern "C" void kernel_launch(void* const* d_in, const int* in_sizes, int n_in,
                              void* d_out, int out_size, void* d_ws, size_t ws_size,
                              hipStream_t stream) {
    const float* x    = (const float*)d_in[0];
    const float* wgt  = (const float*)d_in[1];
    const float* bias = (const float*)d_in[2];
    float* out        = (float*)d_out;

    __bf16* wp = (__bf16*)d_ws;
    __bf16* xp = (__bf16*)((char*)d_ws + XPREP_OFF_B);

    prep_weights<<<dim3((WPREP_ELEMS + 255) / 256), dim3(256), 0, stream>>>(wgt, wp);
    prep_x<<<dim3(8 * 58), dim3(256), 0, stream>>>(x, xp);
    conv3x3_wmma_tdm<<<dim3(8 * 28), dim3(256), SMEM_BYTES, stream>>>(wp, xp, bias, out);
}
